// RingsNet_78529182040898
// MI455X (gfx1250) — matrix operand order repaired and checked by reading, not compiled, not run on hardware
//
#include <hip/hip_runtime.h>
#include <cmath>

typedef __attribute__((ext_vector_type(16))) _Float16 v16h;
typedef __attribute__((ext_vector_type(8)))  _Float16 v8h;
typedef __attribute__((ext_vector_type(8)))  float    v8f;

#define WMMA_F32_F16(a, b, c) \
  __builtin_amdgcn_wmma_f32_16x16x32_f16(false, (a), false, (b), (short)0, (c), false, false)

static constexpr int BT     = 16;    // batch
static constexpr int SEQ    = 512;   // sequence
static constexpr int HIDD   = 256;   // HID
static constexpr int HDIM   = 768;   // H
static constexpr int LAYERS = 12;
static constexpr int NHEADS = 12;
static constexpr int HEADD  = 64;
static constexpr int FFD    = 3072;
static constexpr int TCH    = 128;   // TC

// v16h fragment from two contiguous 8-half runs (LDS or global): 2x *_load_b128.
__device__ inline v16h frag_from_h(const _Float16* p0, const _Float16* p1) {
  v8h lo = *(const v8h*)p0;
  v8h hi = *(const v8h*)p1;
  return __builtin_shufflevector(lo, hi, 0, 1, 2, 3, 4, 5, 6, 7,
                                 8, 9, 10, 11, 12, 13, 14, 15);
}

// ---------------------------------------------------------------------------
// Weight preprocessing: W[K][N] f32 -> WT[N][K] f16, 32x32 LDS tile transpose
// (coalesced on both sides). Amortized once per call; makes every GEMM
// B-fragment a direct K-contiguous global_load_b128 pair.
// ---------------------------------------------------------------------------
__global__ __launch_bounds__(256) void transpose_w_f16(
    const float* __restrict__ in, _Float16* __restrict__ outT, int K, int N)
{
  __shared__ float tile[32][33];
  const int kb = blockIdx.y * 32, nb = blockIdx.x * 32;
#pragma unroll
  for (int i = threadIdx.y; i < 32; i += 8)
    tile[i][threadIdx.x] = in[(size_t)(kb + i) * N + nb + threadIdx.x];
  __syncthreads();
#pragma unroll
  for (int i = threadIdx.y; i < 32; i += 8)
    outT[(size_t)(nb + i) * K + kb + threadIdx.x] = (_Float16)tile[threadIdx.x][i];
}

// Elementwise f32 -> f16 (for the x_t input), 4 elems/thread.
__global__ __launch_bounds__(256) void convert_f16(
    const float* __restrict__ in, _Float16* __restrict__ out, int n)
{
  const int i = (blockIdx.x * 256 + threadIdx.x) * 4;
  if (i + 3 < n) {
    const float4 v = *(const float4*)(in + i);
    out[i + 0] = (_Float16)v.x; out[i + 1] = (_Float16)v.y;
    out[i + 2] = (_Float16)v.z; out[i + 3] = (_Float16)v.w;
  }
}

// ---------------------------------------------------------------------------
// LDS-free f16 GEMM: C = act((A16[M,K] @ BT16[N,K]^T + bias) * oscale).
// M%128==0, N%64==0, K%32==0. Block = 256 threads = 8 wave32s, no barriers:
// per K-step each wave issues 8 global_load_b128 + 4 WMMAs; operand reuse is
// served by WGP$/L2 (f16 weights are near-L2-resident on MI455X's 192MB L2).
// ACT: 0=none, 1=tanh, 2=exact GELU.
// STORET: 1 -> per-batch transposed f16 C[b][n][s] (for V^T).
// OUTF32: 1 -> fp32 output (final projection), else f16.
// ---------------------------------------------------------------------------
template <int ACT, int STORET, int OUTF32>
__global__ __launch_bounds__(256) void gemm16(
    const _Float16* __restrict__ A, const _Float16* __restrict__ Bt,
    const float* __restrict__ bias, void* __restrict__ Cout,
    int M, int N, int K, float oscale)
{
  const int tid  = threadIdx.x;
  const int lane = tid & 31;
  const int wave = tid >> 5;
  const int m0 = blockIdx.y * 128;
  const int n0 = blockIdx.x * 64;
  const int wm = wave & 3;            // M subtiles wm and wm+4
  const int wn = (wave >> 2) << 1;    // N subtiles wn, wn+1
  const int lm = lane & 15;
  const int lh = lane >> 4;

  const _Float16* arow0 = A  + (size_t)(m0 + wm * 16 + lm) * K;
  const _Float16* arow1 = A  + (size_t)(m0 + (wm + 4) * 16 + lm) * K;
  const _Float16* brow0 = Bt + (size_t)(n0 + wn * 16 + lm) * K;
  const _Float16* brow1 = Bt + (size_t)(n0 + (wn + 1) * 16 + lm) * K;

  v8f acc00 = {}, acc01 = {}, acc10 = {}, acc11 = {};

  for (int k0 = 0; k0 < K; k0 += 32) {
    const v16h af0 = frag_from_h(arow0 + k0 + lh * 8, arow0 + k0 + 16 + lh * 8);
    const v16h af1 = frag_from_h(arow1 + k0 + lh * 8, arow1 + k0 + 16 + lh * 8);
    const v16h bf0 = frag_from_h(brow0 + k0 + lh * 8, brow0 + k0 + 16 + lh * 8);
    const v16h bf1 = frag_from_h(brow1 + k0 + lh * 8, brow1 + k0 + 16 + lh * 8);
    if (k0 + 64 < K) {                  // gfx1250 global_prefetch_b8
      __builtin_prefetch(arow0 + k0 + 64, 0, 1);
      __builtin_prefetch(brow0 + k0 + 64, 0, 1);
    }
    acc00 = WMMA_F32_F16(af0, bf0, acc00);
    acc01 = WMMA_F32_F16(af0, bf1, acc01);
    acc10 = WMMA_F32_F16(af1, bf0, acc10);
    acc11 = WMMA_F32_F16(af1, bf1, acc11);
  }

#pragma unroll
  for (int im = 0; im < 2; ++im)
#pragma unroll
    for (int jn = 0; jn < 2; ++jn) {
      v8f acc = (im == 0) ? (jn == 0 ? acc00 : acc01)
                          : (jn == 0 ? acc10 : acc11);
      const int n = n0 + (wn + jn) * 16 + lm;
      const float bv = bias[n];
#pragma unroll
      for (int r = 0; r < 8; ++r) {
        const int m = m0 + (wm + im * 4) * 16 + r + 8 * lh;
        float x = (acc[r] + bv) * oscale;
        if constexpr (ACT == 1) x = tanhf(x);
        if constexpr (ACT == 2) x = 0.5f * x * (1.0f + erff(x * 0.70710678118f));
        if constexpr (OUTF32) {
          ((float*)Cout)[(size_t)m * N + n] = x;
        } else if constexpr (STORET) {
          ((_Float16*)Cout)[((size_t)(m >> 9) * HDIM + n) * SEQ + (m & (SEQ - 1))] =
              (_Float16)x;
        } else {
          ((_Float16*)Cout)[(size_t)m * N + n] = (_Float16)x;
        }
      }
    }
}

// ---------------------------------------------------------------------------
// Flash attention, all-f16 operands: one wave32 per (batch, head, 16-row
// query tile). Q,K f16 row-major [tok,H]; V f16 TRANSPOSED VT[b][feature][s].
// 1/sqrt(d) pre-folded into the Q projection. Every fragment = 2x
// global_load_b128 (f16), softmax uses v_exp_f32. Output f16.
// ---------------------------------------------------------------------------
__global__ __launch_bounds__(32) void flash_attn16(
    const _Float16* __restrict__ Q, const _Float16* __restrict__ K,
    const _Float16* __restrict__ VT, _Float16* __restrict__ O)
{
  __shared__ _Float16 Ps[16][32];   // P bounce: D-layout -> A-layout

  const int lane = threadIdx.x;
  const int lm = lane & 15, lh = lane >> 4;
  const int q0   = blockIdx.x * 16;
  const int head = blockIdx.y;
  const int b    = blockIdx.z;
  const size_t rowbase = (size_t)b * SEQ;
  const int hoff = head * HEADD;

  const _Float16* qp = Q + (rowbase + q0 + lm) * HDIM + hoff;
  v16h aq[2];
#pragma unroll
  for (int c = 0; c < 2; ++c)
    aq[c] = frag_from_h(qp + c * 32 + lh * 8, qp + c * 32 + 16 + lh * 8);

  float mrow[8], lrow[8];
#pragma unroll
  for (int r = 0; r < 8; ++r) { mrow[r] = -1e30f; lrow[r] = 0.0f; }
  v8f oacc[4] = {};

  const _Float16* kb  = K + rowbase * HDIM + hoff;
  const _Float16* vtb = VT + ((size_t)b * HDIM + hoff) * SEQ;

  for (int kt = 0; kt < SEQ; kt += 32) {
    // scores S[16x32] = (Q/sqrt(d)) @ K^T : 4 WMMAs
    v8f s0 = {}, s1 = {};
#pragma unroll
    for (int c = 0; c < 2; ++c) {
      const _Float16* r0 = kb + (size_t)(kt + lm) * HDIM + c * 32 + lh * 8;
      const _Float16* r1 = kb + (size_t)(kt + 16 + lm) * HDIM + c * 32 + lh * 8;
      const v16h bk0 = frag_from_h(r0, r0 + 16);
      const v16h bk1 = frag_from_h(r1, r1 + 16);
      s0 = WMMA_F32_F16(aq[c], bk0, s0);
      s1 = WMMA_F32_F16(aq[c], bk1, s1);
    }

    // online softmax; rows = VGPR idx + lane-half, columns across 16 lanes
#pragma unroll
    for (int r = 0; r < 8; ++r) {
      const float x0 = s0[r], x1 = s1[r];
      float mx = fmaxf(x0, x1);
      mx = fmaxf(mx, __shfl_xor(mx, 1, 32));
      mx = fmaxf(mx, __shfl_xor(mx, 2, 32));
      mx = fmaxf(mx, __shfl_xor(mx, 4, 32));
      mx = fmaxf(mx, __shfl_xor(mx, 8, 32));
      const float mnew  = fmaxf(mrow[r], mx);
      const float alpha = __expf(mrow[r] - mnew);
      const float p0 = __expf(x0 - mnew), p1 = __expf(x1 - mnew);
      float rs = p0 + p1;
      rs += __shfl_xor(rs, 1, 32);
      rs += __shfl_xor(rs, 2, 32);
      rs += __shfl_xor(rs, 4, 32);
      rs += __shfl_xor(rs, 8, 32);
      lrow[r] = lrow[r] * alpha + rs;
      mrow[r] = mnew;
#pragma unroll
      for (int j = 0; j < 4; ++j) oacc[j][r] = oacc[j][r] * alpha;
      Ps[r + 8 * lh][lm]      = (_Float16)p0;
      Ps[r + 8 * lh][16 + lm] = (_Float16)p1;
    }
    __syncthreads();

    const v16h ap = frag_from_h(&Ps[lm][lh * 8], &Ps[lm][16 + lh * 8]);

    // O[16x64] += P[16x32] @ V[32x64] : V^T rows are key-contiguous
#pragma unroll
    for (int j = 0; j < 4; ++j) {
      const _Float16* pv = vtb + (size_t)(j * 16 + lm) * SEQ + kt + lh * 8;
      const v16h bv = frag_from_h(pv, pv + 16);
      oacc[j] = WMMA_F32_F16(ap, bv, oacc[j]);
    }
    __syncthreads();
  }

#pragma unroll
  for (int j = 0; j < 4; ++j)
#pragma unroll
    for (int r = 0; r < 8; ++r) {
      const int row = q0 + r + 8 * lh;
      const int col = hoff + j * 16 + lm;
      O[(rowbase + row) * HDIM + col] = (_Float16)(oacc[j][r] / lrow[r]);
    }
}

// ---------------------------------------------------------------------------
// Timestep embedding: sinusoid(128) -> Linear(256)+SiLU -> Linear(768), f32.
// ---------------------------------------------------------------------------
__global__ __launch_bounds__(256) void time_embed(
    const float* __restrict__ t, const float* __restrict__ t1w,
    const float* __restrict__ t1b, const float* __restrict__ t2w,
    const float* __restrict__ t2b, float* __restrict__ temb)
{
  __shared__ float emb[TCH];
  __shared__ float mid[2 * TCH];
  const int tid = threadIdx.x;
  const int b = blockIdx.x;
  const float tv = t[b];
  if (tid < TCH / 2) {
    const float fr  = __expf(-logf(10000.0f) * (float)tid / (float)(TCH / 2));
    const float arg = tv * fr;
    emb[tid]           = cosf(arg);
    emb[tid + TCH / 2] = sinf(arg);
  }
  __syncthreads();
  float acc = t1b[tid];
  for (int i = 0; i < TCH; ++i) acc += emb[i] * t1w[i * (2 * TCH) + tid];
  mid[tid] = acc / (1.0f + __expf(-acc));   // SiLU
  __syncthreads();
#pragma unroll
  for (int o = 0; o < 3; ++o) {
    const int hcol = tid + o * 256;
    float a2 = t2b[hcol];
    for (int j = 0; j < 2 * TCH; ++j) a2 += mid[j] * t2w[j * HDIM + hcol];
    temb[(size_t)b * HDIM + hcol] = a2;
  }
}

// ---------------------------------------------------------------------------
// LayerNorm kernels (f16 activations, f32 statistics).
// ---------------------------------------------------------------------------
__device__ inline float block_sum256(float v, float* red) {
  const int tid = threadIdx.x;
  red[tid] = v;
  __syncthreads();
  for (int s = 128; s > 0; s >>= 1) {
    if (tid < s) red[tid] += red[tid + s];
    __syncthreads();
  }
  const float r = red[0];
  __syncthreads();
  return r;
}

__global__ __launch_bounds__(256) void add_pos_temb_ln(
    const _Float16* __restrict__ x, const float* __restrict__ pos,
    const float* __restrict__ temb, const float* __restrict__ w,
    const float* __restrict__ bsh, _Float16* __restrict__ out)
{
  __shared__ float red[256];
  const int tid = threadIdx.x;
  const int row = blockIdx.x;
  const int bb = row / SEQ, s = row % SEQ;
  float vals[3];
  float sum = 0.0f;
#pragma unroll
  for (int o = 0; o < 3; ++o) {
    const int c = tid + o * 256;
    vals[o] = (float)x[(size_t)row * HDIM + c] + pos[(size_t)s * HDIM + c] +
              temb[(size_t)bb * HDIM + c];
    sum += vals[o];
  }
  const float mean = block_sum256(sum, red) / (float)HDIM;
  float sq = 0.0f;
#pragma unroll
  for (int o = 0; o < 3; ++o) { const float d = vals[o] - mean; sq += d * d; }
  const float var = block_sum256(sq, red) / (float)HDIM;
  const float inv = rsqrtf(var + 1e-12f);
#pragma unroll
  for (int o = 0; o < 3; ++o) {
    const int c = tid + o * 256;
    out[(size_t)row * HDIM + c] = (_Float16)((vals[o] - mean) * inv * w[c] + bsh[c]);
  }
}

__global__ __launch_bounds__(256) void residual_ln(
    const _Float16* __restrict__ resid, const _Float16* __restrict__ delta,
    const float* __restrict__ w, const float* __restrict__ bsh,
    _Float16* __restrict__ out)
{
  __shared__ float red[256];
  const int tid = threadIdx.x;
  const int row = blockIdx.x;
  float vals[3];
  float sum = 0.0f;
#pragma unroll
  for (int o = 0; o < 3; ++o) {
    const int c = tid + o * 256;
    vals[o] = (float)resid[(size_t)row * HDIM + c] +
              (float)delta[(size_t)row * HDIM + c];
    sum += vals[o];
  }
  const float mean = block_sum256(sum, red) / (float)HDIM;
  float sq = 0.0f;
#pragma unroll
  for (int o = 0; o < 3; ++o) { const float d = vals[o] - mean; sq += d * d; }
  const float var = block_sum256(sq, red) / (float)HDIM;
  const float inv = rsqrtf(var + 1e-12f);
#pragma unroll
  for (int o = 0; o < 3; ++o) {
    const int c = tid + o * 256;
    out[(size_t)row * HDIM + c] = (_Float16)((vals[o] - mean) * inv * w[c] + bsh[c]);
  }
}

// ---------------------------------------------------------------------------
static inline void launch_gemm(const _Float16* A, const _Float16* Bt,
                               const float* bias, void* C, int M, int N, int K,
                               int act, int storeT, int outF32, float oscale,
                               hipStream_t stream)
{
  const dim3 grid((unsigned)(N / 64), (unsigned)(M / 128));
  const dim3 blk(256);
  if (outF32)        gemm16<0, 0, 1><<<grid, blk, 0, stream>>>(A, Bt, bias, C, M, N, K, oscale);
  else if (storeT)   gemm16<0, 1, 0><<<grid, blk, 0, stream>>>(A, Bt, bias, C, M, N, K, oscale);
  else if (act == 1) gemm16<1, 0, 0><<<grid, blk, 0, stream>>>(A, Bt, bias, C, M, N, K, oscale);
  else if (act == 2) gemm16<2, 0, 0><<<grid, blk, 0, stream>>>(A, Bt, bias, C, M, N, K, oscale);
  else               gemm16<0, 0, 0><<<grid, blk, 0, stream>>>(A, Bt, bias, C, M, N, K, oscale);
}

static inline void launch_transpose(const float* W, _Float16* WT, int K, int N,
                                    hipStream_t stream)
{
  transpose_w_f16<<<dim3((unsigned)(N / 32), (unsigned)(K / 32)), dim3(32, 8), 0,
                    stream>>>(W, WT, K, N);
}

extern "C" void kernel_launch(void* const* d_in, const int* in_sizes, int n_in,
                              void* d_out, int out_size, void* d_ws, size_t ws_size,
                              hipStream_t stream)
{
  const float* x_t   = (const float*)d_in[0];
  const float* t     = (const float*)d_in[1];
  const float* up1w  = (const float*)d_in[2];
  const float* up1b  = (const float*)d_in[3];
  const float* up2w  = (const float*)d_in[4];
  const float* up2b  = (const float*)d_in[5];
  const float* t1w   = (const float*)d_in[6];
  const float* t1b   = (const float*)d_in[7];
  const float* t2w   = (const float*)d_in[8];
  const float* t2b   = (const float*)d_in[9];
  const float* pos   = (const float*)d_in[10];
  const float* lnw   = (const float*)d_in[11];
  const float* lnb   = (const float*)d_in[12];
  const float* qw    = (const float*)d_in[13];
  const float* qb    = (const float*)d_in[14];
  const float* kw    = (const float*)d_in[15];
  const float* kbb   = (const float*)d_in[16];
  const float* vw    = (const float*)d_in[17];
  const float* vbb   = (const float*)d_in[18];
  const float* aow   = (const float*)d_in[19];
  const float* aob   = (const float*)d_in[20];
  const float* alnw  = (const float*)d_in[21];
  const float* alnb  = (const float*)d_in[22];
  const float* iw    = (const float*)d_in[23];
  const float* ib    = (const float*)d_in[24];
  const float* ow    = (const float*)d_in[25];
  const float* ob    = (const float*)d_in[26];
  const float* olnw  = (const float*)d_in[27];
  const float* olnb  = (const float*)d_in[28];
  const float* d1w   = (const float*)d_in[29];
  const float* d1b   = (const float*)d_in[30];
  const float* d2w   = (const float*)d_in[31];
  const float* d2b   = (const float*)d_in[32];
  float* out = (float*)d_out;

  const size_t tok = (size_t)BT * SEQ;   // 8192 tokens
  const size_t HH  = (size_t)HDIM * HDIM;

  // Workspace layout: f32 temb, then f16 activations, then f16 weights.
  float* temb = (float*)d_ws;                          // [16, 768]
  _Float16* cur = (_Float16*)(temb + (size_t)BT * HDIM);
  auto alloc = [&](size_t n) { _Float16* p = cur; cur += n; return p; };
  _Float16* h16    = alloc(tok * HDIM);
  _Float16* q16    = alloc(tok * HDIM);
  _Float16* k16    = alloc(tok * HDIM);
  _Float16* vt16   = alloc(tok * HDIM);   // [16, 768, 512] V transposed
  _Float16* ctx16  = alloc(tok * HDIM);
  _Float16* proj16 = alloc(tok * HDIM);
  _Float16* ffb16  = alloc(tok * FFD);
  _Float16* x16    = alloc(tok * 2 * HIDD);
  _Float16* wup1   = alloc((size_t)2 * HIDD * HDIM);
  _Float16* wup2   = alloc(HH);
  _Float16* wq     = alloc(HH * LAYERS);
  _Float16* wk     = alloc(HH * LAYERS);
  _Float16* wv     = alloc(HH * LAYERS);
  _Float16* wao    = alloc(HH * LAYERS);
  _Float16* wi     = alloc((size_t)HDIM * FFD * LAYERS);
  _Float16* wo     = alloc((size_t)HDIM * FFD * LAYERS);
  _Float16* wd1    = alloc(HH);
  _Float16* wd2    = alloc((size_t)HDIM * HIDD);

  const dim3 blk(256);
  const int M = (int)tok;

  // ---- One-time (per call) weight convert+transpose to f16 ----
  launch_transpose(up1w, wup1, 2 * HIDD, HDIM, stream);
  launch_transpose(up2w, wup2, HDIM, HDIM, stream);
  for (int i = 0; i < LAYERS; ++i) {
    launch_transpose(qw + i * HH, wq + i * HH, HDIM, HDIM, stream);
    launch_transpose(kw + i * HH, wk + i * HH, HDIM, HDIM, stream);
    launch_transpose(vw + i * HH, wv + i * HH, HDIM, HDIM, stream);
    launch_transpose(aow + i * HH, wao + i * HH, HDIM, HDIM, stream);
    launch_transpose(iw + (size_t)i * HDIM * FFD, wi + (size_t)i * HDIM * FFD,
                     HDIM, FFD, stream);
    launch_transpose(ow + (size_t)i * HDIM * FFD, wo + (size_t)i * HDIM * FFD,
                     FFD, HDIM, stream);
  }
  launch_transpose(d1w, wd1, HDIM, HDIM, stream);
  launch_transpose(d2w, wd2, HDIM, HIDD, stream);
  convert_f16<<<dim3((unsigned)(tok * 2 * HIDD / 1024)), blk, 0, stream>>>(
      x_t, x16, (int)(tok * 2 * HIDD));

  // ---- Embedding path ----
  time_embed<<<dim3(BT), blk, 0, stream>>>(t, t1w, t1b, t2w, t2b, temb);
  launch_gemm(x16, wup1, up1b, ffb16, M, HDIM, 2 * HIDD, 1, 0, 0, 1.0f, stream);
  launch_gemm(ffb16, wup2, up2b, ctx16, M, HDIM, HDIM, 0, 0, 0, 1.0f, stream);
  add_pos_temb_ln<<<dim3((unsigned)tok), blk, 0, stream>>>(ctx16, pos, temb, lnw, lnb, h16);

  // ---- Encoder layers ----
  for (int i = 0; i < LAYERS; ++i) {
    launch_gemm(h16, wq + i * HH, qb + (size_t)i * HDIM, q16, M, HDIM, HDIM,
                0, 0, 0, 0.125f, stream);                       // scale folded
    launch_gemm(h16, wk + i * HH, kbb + (size_t)i * HDIM, k16, M, HDIM, HDIM,
                0, 0, 0, 1.0f, stream);
    launch_gemm(h16, wv + i * HH, vbb + (size_t)i * HDIM, vt16, M, HDIM, HDIM,
                0, 1, 0, 1.0f, stream);                         // store V^T
    flash_attn16<<<dim3(SEQ / 16, NHEADS, BT), dim3(32), 0, stream>>>(
        q16, k16, vt16, ctx16);
    launch_gemm(ctx16, wao + i * HH, aob + (size_t)i * HDIM, proj16, M, HDIM,
                HDIM, 0, 0, 0, 1.0f, stream);
    residual_ln<<<dim3((unsigned)tok), blk, 0, stream>>>(
        h16, proj16, alnw + (size_t)i * HDIM, alnb + (size_t)i * HDIM, h16);
    launch_gemm(h16, wi + (size_t)i * HDIM * FFD, ib + (size_t)i * FFD, ffb16,
                M, FFD, HDIM, 2, 0, 0, 1.0f, stream);           // exact GELU
    launch_gemm(ffb16, wo + (size_t)i * HDIM * FFD, ob + (size_t)i * HDIM,
                proj16, M, HDIM, FFD, 0, 0, 0, 1.0f, stream);
    residual_ln<<<dim3((unsigned)tok), blk, 0, stream>>>(
        h16, proj16, olnw + (size_t)i * HDIM, olnb + (size_t)i * HDIM, h16);
  }

  // ---- Output down-projection ----
  launch_gemm(h16, wd1, d1b, proj16, M, HDIM, HDIM, 1, 0, 0, 1.0f, stream);
  launch_gemm(proj16, wd2, d2b, out, M, HIDD, HDIM, 0, 0, 1, 1.0f, stream);

  (void)in_sizes; (void)n_in; (void)out_size; (void)ws_size;
}
